// graph_encode_12773232738895
// MI455X (gfx1250) — matrix operand order repaired and checked by reading, not compiled, not run on hardware
//
#include <hip/hip_runtime.h>

// ---------------- fixed problem dims ----------------
#define B_   4
#define NE_  1536
#define NR_  512
#define NN   2048          // NE_ + NR_
#define D_   128
#define H_   4
#define DH_  32
#define P_   2
#define M_   (B_ * NN)     // 8192 rows when flattened [B*N, D]
#define FF   512           // 4*D
#define SCALE 0.17677669529663687f   // 1/sqrt(32)

typedef __bf16 bf16;
typedef __attribute__((ext_vector_type(16))) __bf16 v16bf;
typedef __attribute__((ext_vector_type(8)))  __bf16 v8bf;
typedef __attribute__((ext_vector_type(8)))  float  v8f;
typedef __attribute__((ext_vector_type(4)))  unsigned int u32x4;
typedef __attribute__((ext_vector_type(8)))  int i32x8;
typedef __attribute__((ext_vector_type(4)))  int i32x4;

#if __has_builtin(__builtin_amdgcn_tensor_load_to_lds) && __has_builtin(__builtin_amdgcn_s_wait_tensorcnt)
#define HAVE_TDM 1
#else
#define HAVE_TDM 0
#endif

// ---------------- WMMA helpers ----------------
__device__ inline v8f wmma_bf16(v16bf a, v16bf b, v8f c) {
    // D = A(16x32 bf16) * B(32x16 bf16) + C(16x16 f32)
    return __builtin_amdgcn_wmma_f32_16x16x32_bf16(
        /*neg_a=*/false, a, /*neg_b=*/false, b,
        /*c_mod=*/(short)0, c, /*reuse_a=*/false, /*reuse_b=*/false);
}

// A-operand (16x32, MxK) loader. rowbase points at this lane's row start + k0.
// Layout: lane l holds row (l&15); lanes<16 take K {0..15}, lanes>=16 take {16..31}.
__device__ inline v16bf load_a_tile(const bf16* rowbase, int lane) {
    int kb = (lane < 16) ? 0 : 16;
    v8bf x = *(const v8bf*)(rowbase + kb);
    v8bf y = *(const v8bf*)(rowbase + kb + 8);
    v16bf a;
#pragma unroll
    for (int i = 0; i < 8; ++i) { a[i] = x[i]; a[8 + i] = y[i]; }
    return a;
}

// B-operand (32x16, KxN) loader: per lane, 16 contiguous bf16 (this lane's
// column; K-slice 0..15 / 16..31 already selected by caller via +0/+16 offset).
__device__ inline v16bf load_b_tile(const bf16* colbase) {
    v8bf x = *(const v8bf*)(colbase);
    v8bf y = *(const v8bf*)(colbase + 8);
    v16bf b;
#pragma unroll
    for (int i = 0; i < 8; ++i) { b[i] = x[i]; b[8 + i] = y[i]; }
    return b;
}

#if HAVE_TDM
// Issue one TDM DMA: global (2-byte elements) -> LDS. 1D tile if tile1==0, else
// 2D tile (tile0 wide, tile1 rows, row stride stride0). All args wave-uniform.
__device__ inline void tdm_load(unsigned int lds_off, const void* gptr,
                                unsigned int dim0, unsigned int dim1,
                                unsigned int tile0, unsigned int tile1,
                                unsigned int stride0) {
    unsigned long long ga = (unsigned long long)gptr;
    u32x4 g0;
    g0[0] = 1u;                                                            // count=1, user desc
    g0[1] = (unsigned int)__builtin_amdgcn_readfirstlane((int)lds_off);    // lds_addr
    g0[2] = (unsigned int)__builtin_amdgcn_readfirstlane((int)(unsigned int)ga);
    g0[3] = (unsigned int)__builtin_amdgcn_readfirstlane(
                (int)(((unsigned int)(ga >> 32) & 0x01FFFFFFu) | (2u << 30))); // addr hi | type=2
    i32x8 g1;
    g1[0] = 0x00010000;                        // workgroup_mask=0, data_size=1 (2 bytes)
    g1[1] = (int)(dim0 << 16);                 // tensor_dim0[15:0] @ bits 63:48
    g1[2] = (int)((dim0 >> 16) | (dim1 << 16));// tensor_dim0[31:16] | tensor_dim1[15:0]
    g1[3] = (int)(tile0 << 16);                // tile_dim0 @ bits 127:112
    g1[4] = (int)tile1;                        // tile_dim1 (0 = 1D)
    g1[5] = (int)stride0;                      // tensor_dim0_stride[31:0]
    g1[6] = 0;
    g1[7] = 0;
    i32x4 gz = {0, 0, 0, 0};
#if defined(__clang_major__) && __clang_major__ >= 23
    i32x8 gz8 = {0, 0, 0, 0, 0, 0, 0, 0};
    __builtin_amdgcn_tensor_load_to_lds(g0, g1, gz, gz, gz8, 0);
#else
    __builtin_amdgcn_tensor_load_to_lds(g0, g1, gz, gz, 0);
#endif
}
#endif

// ---------------- kernel 1: build vgraph (concat + embedding gather) ----------------
__global__ __launch_bounds__(256) void build_vgraph_kernel(
    const float* __restrict__ vents, const int* __restrict__ rels,
    const float* __restrict__ renc_w, float* __restrict__ vg32,
    bf16* __restrict__ vg16)
{
    int idx = blockIdx.x * 256 + threadIdx.x;        // over B*N*D
    int d = idx & (D_ - 1);
    int n = (idx >> 7) & (NN - 1);
    int b = idx >> 18;                               // / (NN*D_)
    float v;
    if (n < NE_) v = vents[((size_t)b * NE_ + n) * D_ + d];
    else         v = renc_w[(size_t)rels[b * NR_ + (n - NE_)] * D_ + d];
    vg32[idx] = v;
    vg16[idx] = (bf16)v;
}

// ---------------- kernel 2: weight transpose+convert: in f32 [K,N] -> out bf16 [N,K] ----------------
__global__ __launch_bounds__(256) void transpose_bf16_kernel(
    const float* __restrict__ in, bf16* __restrict__ out, int K, int Nc)
{
    int idx = blockIdx.x * 256 + threadIdx.x;
    if (idx >= K * Nc) return;
    int k = idx / Nc, n = idx - k * Nc;
    out[(size_t)n * K + k] = (bf16)in[idx];
}

// ---------------- kernel 3: wave-tile GEMM, 1x4 register blocking ----------------
// A: [M,K] bf16 row-major.  WT: [Nc,K] bf16 (pre-transposed).
// One wave computes a 16x64 strip (4 accumulators) so each A-tile feeds 4 WMMAs.
#define OUT_F32     0
#define OUT_BF16    1
#define OUT_QK_HEAD 2   // scatter to [B,H,N,DH]
#define OUT_V_TRANS 3   // scatter to [B,H,DH,N]
__global__ __launch_bounds__(128) void gemm16_kernel(
    const bf16* __restrict__ A, const bf16* __restrict__ WT,
    const float* __restrict__ bias, const float* __restrict__ prelu,
    void* __restrict__ out, int M, int K, int Nc, int mode)
{
    int lane = threadIdx.x & 31;
    int wave = blockIdx.x * 4 + (threadIdx.x >> 5);
    int tilesN4 = Nc >> 6;                 // groups of 4 16-wide tiles
    int m0 = (wave / tilesN4) << 4;
    int n0 = (wave - (wave / tilesN4) * tilesN4) << 6;
    if (m0 >= M) return;

    int l15 = lane & 15;
    int hi16 = (lane < 16) ? 0 : 16;
    const bf16* arow = A + (size_t)(m0 + l15) * K;
    const bf16* w0 = WT + (size_t)(n0 + l15) * K + hi16;
    const bf16* w1 = w0 + (size_t)16 * K;
    const bf16* w2 = w0 + (size_t)32 * K;
    const bf16* w3 = w0 + (size_t)48 * K;

    v8f acc[4];
#pragma unroll
    for (int t = 0; t < 4; ++t) acc[t] = (v8f){};
    for (int k0 = 0; k0 < K; k0 += 32) {
        v16bf a = load_a_tile(arow + k0, lane);        // loaded once, used 4x
        acc[0] = wmma_bf16(a, load_b_tile(w0 + k0), acc[0]);
        acc[1] = wmma_bf16(a, load_b_tile(w1 + k0), acc[1]);
        acc[2] = wmma_bf16(a, load_b_tile(w2 + k0), acc[2]);
        acc[3] = wmma_bf16(a, load_b_tile(w3 + k0), acc[3]);
    }

    int rbase = m0 + ((lane < 16) ? 0 : 8);
#pragma unroll
    for (int t = 0; t < 4; ++t) {
        int col = n0 + t * 16 + l15;
        float bcol = bias ? bias[col] : 0.0f;
        float pcol = prelu ? prelu[col] : 0.0f;
#pragma unroll
        for (int r = 0; r < 8; ++r) {
            int row = rbase + r;
            float v = acc[t][r] + bcol;
            if (prelu) v = (v >= 0.0f) ? v : pcol * v;
            if (mode == OUT_F32) {
                ((float*)out)[(size_t)row * Nc + col] = v;
            } else if (mode == OUT_BF16) {
                ((bf16*)out)[(size_t)row * Nc + col] = (bf16)v;
            } else {
                int b = row >> 11;            // row / NN
                int node = row & (NN - 1);
                int h = col >> 5, d = col & 31;
                if (mode == OUT_QK_HEAD)
                    ((bf16*)out)[((((size_t)b * H_ + h) * NN + node) << 5) + d] = (bf16)v;
                else
                    ((bf16*)out)[((((size_t)b * H_ + h) << 5) + d) * NN + node] = (bf16)v;
            }
        }
    }
}

// ---------------- kernel 4: fused masked attention ----------------
// Flash-style, 1 wave per 16-row query block. K/V tiles staged into LDS by the
// Tensor Data Mover (double-buffered, s_wait_tensorcnt), probs re-laid-out
// through LDS for the PV WMMA.
// Q,K: [B,H,N,DH] bf16   V: [B,H,DH,N] bf16   adj: [B,N,N] i32   O: [B*N,D] bf16
__global__ __launch_bounds__(128) void attn_kernel(
    const bf16* __restrict__ Qh, const bf16* __restrict__ Kh,
    const bf16* __restrict__ Vt, const int* __restrict__ adj,
    bf16* __restrict__ Obuf)
{
#if HAVE_TDM
    __shared__ __align__(16) bf16 ldsK[4][2][32 * 32];   // 2KB tiles, ping-pong
    __shared__ __align__(16) bf16 ldsV[4][2][32 * 32];
#endif
    __shared__ __align__(16) bf16 ldsP[4][16 * 32];      // probs staging
    int lane = threadIdx.x & 31;
    int w = threadIdx.x >> 5;
    int gw = blockIdx.x * 4 + w;          // 0 .. B*H*(NN/16)-1 = 2047
    int iblk = gw & 127;                  // NN/16 = 128
    int bh = gw >> 7;
    int b = bh >> 2, h = bh & 3;
    int i0 = iblk << 4;
    int l15 = lane & 15;
    int hi = lane >> 4;                   // lane group: 0 or 1
    bf16* myp = ldsP[w];

    // Q tile (reused across all 64 K/V chunks)
    const bf16* qbase = Qh + (((size_t)(b * H_ + h) * NN + i0 + l15) << 5);
    v16bf qa = load_a_tile(qbase, lane);

    const bf16* kgb = Kh + (((size_t)(b * H_ + h) * NN) << 5);   // chunk c: + c*1024
    const bf16* vgb = Vt + ((size_t)(b * H_ + h) << 5) * NN;     // chunk c: + c*32
    const int* abase = adj + ((size_t)b * NN + i0 + hi * 8) * NN;

    v8f o0 = {}, o1 = {};
    float rmax[8], rsum[8];
#pragma unroll
    for (int r = 0; r < 8; ++r) { rmax[r] = -3.0e38f; rsum[r] = 0.0f; }

#if HAVE_TDM
    unsigned int koff[2] = { (unsigned int)(size_t)&ldsK[w][0][0],
                             (unsigned int)(size_t)&ldsK[w][1][0] };
    unsigned int voff[2] = { (unsigned int)(size_t)&ldsV[w][0][0],
                             (unsigned int)(size_t)&ldsV[w][1][0] };
    // prologue: DMA chunk 0 into buffer 0
    tdm_load(koff[0], kgb, /*dim0*/1024, 0, /*tile0*/1024, 0, 1024);          // 1D 2KB
    tdm_load(voff[0], vgb, /*dim0*/NN, 32, /*tile0*/32, /*tile1*/32, NN);     // 32x32 strided
#endif

    for (int c = 0; c < NN / 32; ++c) {
        int j0 = c * 32;
#if HAVE_TDM
        int buf = c & 1;
        if (c + 1 < NN / 32) {
            tdm_load(koff[buf ^ 1], kgb + (size_t)(c + 1) * 1024, 1024, 0, 1024, 0, 1024);
            tdm_load(voff[buf ^ 1], vgb + (size_t)(c + 1) * 32, NN, 32, 32, 32, NN);
            __builtin_amdgcn_s_wait_tensorcnt(2);   // chunk c's two DMAs done
        } else {
            __builtin_amdgcn_s_wait_tensorcnt(0);
        }
        const bf16* kt = ldsK[w][buf];   // row = j-local, 32 bf16 per row
        const bf16* vt = ldsV[w][buf];   // row = d,       32 bf16 per row
        v16bf kb0 = load_b_tile(kt + (size_t)l15 * 32 + hi * 16);
        v16bf kb1 = load_b_tile(kt + (size_t)(16 + l15) * 32 + hi * 16);
#else
        v16bf kb0 = load_b_tile(kgb + (size_t)(j0 + l15) * 32 + hi * 16);
        v16bf kb1 = load_b_tile(kgb + (size_t)(j0 + 16 + l15) * 32 + hi * 16);
#endif
        if (j0 + 32 < NN)   // adj is the only HBM-heavy stream: stay ahead of it
            __builtin_prefetch(abase + j0 + 32, 0, 3);

        v8f z = {};
        v8f s0 = wmma_bf16(qa, kb0, z);
        v8f s1 = wmma_bf16(qa, kb1, z);

        // mask + scale + online softmax (row stats per lane-group, per C-layout row)
#pragma unroll
        for (int r = 0; r < 8; ++r) {
            const int* ar = abase + (size_t)r * NN + j0;
            float x0 = ar[l15]      ? s0[r] * SCALE : -1e9f;
            float x1 = ar[16 + l15] ? s1[r] * SCALE : -1e9f;
            float m = fmaxf(x0, x1);
            m = fmaxf(m, __shfl_xor(m, 1));
            m = fmaxf(m, __shfl_xor(m, 2));
            m = fmaxf(m, __shfl_xor(m, 4));
            m = fmaxf(m, __shfl_xor(m, 8));
            float nm = fmaxf(rmax[r], m);
            float corr = __expf(rmax[r] - nm);
            float p0 = __expf(x0 - nm);
            float p1 = __expf(x1 - nm);
            float ps = p0 + p1;
            ps += __shfl_xor(ps, 1);
            ps += __shfl_xor(ps, 2);
            ps += __shfl_xor(ps, 4);
            ps += __shfl_xor(ps, 8);
            rsum[r] = rsum[r] * corr + ps;
            rmax[r] = nm;
            o0[r] *= corr;
            o1[r] *= corr;
            int rowl = r + hi * 8;
            myp[rowl * 32 + l15]      = (bf16)p0;   // C-layout -> LDS row-major
            myp[rowl * 32 + 16 + l15] = (bf16)p1;
        }

        // re-read probs in A-operand layout (same wave; compiler inserts s_wait_dscnt)
        v16bf pa = load_a_tile(myp + l15 * 32, lane);
#if HAVE_TDM
        v16bf vb0 = load_b_tile(vt + (size_t)l15 * 32 + hi * 16);        // d = 0..15
        v16bf vb1 = load_b_tile(vt + (size_t)(16 + l15) * 32 + hi * 16); // d = 16..31
#else
        v16bf vb0 = load_b_tile(vgb + (size_t)l15 * NN + j0 + hi * 16);
        v16bf vb1 = load_b_tile(vgb + (size_t)(16 + l15) * NN + j0 + hi * 16);
#endif
        o0 = wmma_bf16(pa, vb0, o0);
        o1 = wmma_bf16(pa, vb1, o1);
    }

#pragma unroll
    for (int r = 0; r < 8; ++r) {
        float inv = 1.0f / rsum[r];
        int row_i = i0 + r + hi * 8;
        size_t o = ((size_t)b * NN + row_i) * D_ + h * DH_;
        Obuf[o + l15]      = (bf16)(o0[r] * inv);
        Obuf[o + 16 + l15] = (bf16)(o1[r] * inv);
    }
}

// ---------------- kernel 5: LayerNorm over D (optional residual), f32+bf16 outputs ----------------
__global__ __launch_bounds__(128) void ln_kernel(
    const float* __restrict__ x, const float* __restrict__ res,
    const float* __restrict__ g, const float* __restrict__ bta,
    float* __restrict__ y32, bf16* __restrict__ y16)
{
    int row = blockIdx.x, t = threadIdx.x;
    float v = x[(size_t)row * D_ + t];
    if (res) v += res[(size_t)row * D_ + t];
    __shared__ float sm[4], sv[4];
    float s = v;
    s += __shfl_xor(s, 1);  s += __shfl_xor(s, 2);
    s += __shfl_xor(s, 4);  s += __shfl_xor(s, 8);  s += __shfl_xor(s, 16);
    if ((t & 31) == 0) sm[t >> 5] = s;
    __syncthreads();
    float mean = (sm[0] + sm[1] + sm[2] + sm[3]) * (1.0f / D_);
    float dv = v - mean;
    float s2 = dv * dv;
    s2 += __shfl_xor(s2, 1); s2 += __shfl_xor(s2, 2);
    s2 += __shfl_xor(s2, 4); s2 += __shfl_xor(s2, 8); s2 += __shfl_xor(s2, 16);
    if ((t & 31) == 0) sv[t >> 5] = s2;
    __syncthreads();
    float var = (sv[0] + sv[1] + sv[2] + sv[3]) * (1.0f / D_);
    float o = dv * rsqrtf(var + 1e-5f) * g[t] + bta[t];
    y32[(size_t)row * D_ + t] = o;
    y16[(size_t)row * D_ + t] = (bf16)o;
}

// ---------------- kernel 6: pack outputs (glob, gents, emask) ----------------
__global__ __launch_bounds__(256) void finalize_kernel(
    const float* __restrict__ vg, const int* __restrict__ entlens,
    float* __restrict__ out, int out_size)
{
    int idx = blockIdx.x * 256 + threadIdx.x;
    if (idx >= out_size) return;
    const int globN = B_ * D_;
    const int gentsN = M_ * D_;
    if (idx < globN) {
        int b = idx >> 7, d = idx & (D_ - 1);
        out[idx] = vg[((size_t)b * NN + entlens[b]) * D_ + d];
    } else if (idx < globN + gentsN) {
        out[idx] = vg[idx - globN];
    } else {
        out[idx] = 1.0f;   // emask is all-true
    }
}

// ---------------- host orchestration ----------------
extern "C" void kernel_launch(void* const* d_in, const int* in_sizes, int n_in,
                              void* d_out, int out_size, void* d_ws, size_t ws_size,
                              hipStream_t stream) {
    (void)in_sizes; (void)n_in; (void)ws_size;
    const float* vents   = (const float*)d_in[0];
    const int*   rels    = (const int*)d_in[1];
    const int*   adj     = (const int*)d_in[2];
    const int*   entlens = (const int*)d_in[3];
    const float* renc_w  = (const float*)d_in[4];
    const float* Wq = (const float*)d_in[5];   const float* bq = (const float*)d_in[6];
    const float* Wk = (const float*)d_in[7];   const float* bk = (const float*)d_in[8];
    const float* Wv = (const float*)d_in[9];   const float* bv = (const float*)d_in[10];
    const float* Wo = (const float*)d_in[11];  const float* bo = (const float*)d_in[12];
    const float* l1w = (const float*)d_in[13]; const float* l1b = (const float*)d_in[14];
    const float* l2w = (const float*)d_in[15]; const float* l2b = (const float*)d_in[16];
    const float* lng = (const float*)d_in[17]; const float* lnb = (const float*)d_in[18];
    const float* pre = (const float*)d_in[19];

    // workspace carve-up (256B aligned)
    char* ws = (char*)d_ws;
    size_t off = 0;
    auto alloc = [&](size_t bytes) { size_t o = off; off += (bytes + 255) & ~(size_t)255; return o; };
    float* vg32 = (float*)(ws + alloc((size_t)M_ * D_ * 4));
    bf16*  vg16 = (bf16*) (ws + alloc((size_t)M_ * D_ * 2));
    bf16*  Qh   = (bf16*) (ws + alloc((size_t)M_ * D_ * 2));
    bf16*  Kh   = (bf16*) (ws + alloc((size_t)M_ * D_ * 2));
    bf16*  Vt   = (bf16*) (ws + alloc((size_t)M_ * D_ * 2));
    bf16*  Ob   = (bf16*) (ws + alloc((size_t)M_ * D_ * 2));
    float* X1   = (float*)(ws + alloc((size_t)M_ * D_ * 4));
    float* t32  = (float*)(ws + alloc((size_t)M_ * D_ * 4));
    bf16*  t16  = (bf16*) (ws + alloc((size_t)M_ * D_ * 2));
    bf16*  h16  = (bf16*) (ws + alloc((size_t)M_ * FF * 2));
    float* q2   = (float*)(ws + alloc((size_t)M_ * D_ * 4));
    bf16*  WqT  = (bf16*) (ws + alloc((size_t)D_ * D_ * 2));
    bf16*  WkT  = (bf16*) (ws + alloc((size_t)D_ * D_ * 2));
    bf16*  WvT  = (bf16*) (ws + alloc((size_t)D_ * D_ * 2));
    bf16*  WoT  = (bf16*) (ws + alloc((size_t)D_ * D_ * 2));
    bf16*  L1T  = (bf16*) (ws + alloc((size_t)FF * D_ * 2));
    bf16*  L2T  = (bf16*) (ws + alloc((size_t)D_ * FF * 2));

    build_vgraph_kernel<<<(M_ * D_) / 256, 256, 0, stream>>>(vents, rels, renc_w, vg32, vg16);

    // 1 wave per 16x64 strip -> (M/16)*(Nc/64) waves, 4 waves/block
    const int gemmDD_blocks = (M_ / 16) * (D_ / 64) / 4;   // 256
    const int gemmFF_blocks = (M_ / 16) * (FF / 64) / 4;   // 1024
    const int trDD = (D_ * D_) / 256, trDF = (D_ * FF) / 256;

    for (int p = 0; p < P_; ++p) {
        // weight transposes (f32 -> bf16 [N,K])
        transpose_bf16_kernel<<<trDD, 256, 0, stream>>>(Wq  + (size_t)p * D_ * D_, WqT, D_, D_);
        transpose_bf16_kernel<<<trDD, 256, 0, stream>>>(Wk  + (size_t)p * D_ * D_, WkT, D_, D_);
        transpose_bf16_kernel<<<trDD, 256, 0, stream>>>(Wv  + (size_t)p * D_ * D_, WvT, D_, D_);
        transpose_bf16_kernel<<<trDD, 256, 0, stream>>>(Wo  + (size_t)p * D_ * D_, WoT, D_, D_);
        transpose_bf16_kernel<<<trDF, 256, 0, stream>>>(l1w + (size_t)p * D_ * FF, L1T, D_, FF);
        transpose_bf16_kernel<<<trDF, 256, 0, stream>>>(l2w + (size_t)p * FF * D_, L2T, FF, D_);

        // Q/K/V projections with head-layout scatter epilogues
        gemm16_kernel<<<gemmDD_blocks, 128, 0, stream>>>(vg16, WqT, bq + p * D_, nullptr, Qh, M_, D_, D_, OUT_QK_HEAD);
        gemm16_kernel<<<gemmDD_blocks, 128, 0, stream>>>(vg16, WkT, bk + p * D_, nullptr, Kh, M_, D_, D_, OUT_QK_HEAD);
        gemm16_kernel<<<gemmDD_blocks, 128, 0, stream>>>(vg16, WvT, bv + p * D_, nullptr, Vt, M_, D_, D_, OUT_V_TRANS);

        // fused masked attention: B*H*(NN/16) = 2048 waves, 4 waves/block
        attn_kernel<<<512, 128, 0, stream>>>(Qh, Kh, Vt, adj, Ob);

        // output projection + Block FFN
        gemm16_kernel<<<gemmDD_blocks, 128, 0, stream>>>(Ob, WoT, bo + p * D_, nullptr, X1, M_, D_, D_, OUT_F32);
        ln_kernel<<<M_, 128, 0, stream>>>(X1, nullptr, lng + p * D_, lnb + p * D_, t32, t16);
        gemm16_kernel<<<gemmFF_blocks, 128, 0, stream>>>(t16, L1T, l1b + p * FF, pre + p * FF, h16, M_, D_, FF, OUT_BF16);
        gemm16_kernel<<<gemmDD_blocks, 128, 0, stream>>>(h16, L2T, l2b + p * D_, nullptr, q2, M_, FF, D_, OUT_F32);
        ln_kernel<<<M_, 128, 0, stream>>>(q2, t32, lng + p * D_, lnb + p * D_, vg32, vg16);
    }

    finalize_kernel<<<(out_size + 255) / 256, 256, 0, stream>>>(vg32, entlens, (float*)d_out, out_size);
}
